// DeciLMSSMDecoderLayer_59725815218320
// MI455X (gfx1250) — compile-verified, matrix operand order checked by reading, not run on hardware
//
#include <hip/hip_runtime.h>
#include <hip/hip_bf16.h>
#include <stdint.h>

// ---------------- model constants ----------------
#define H_        4096
#define DS_       4096
#define S_        128
#define G_        1
#define NH_       64
#define HD_       64
#define KCONV_    4
#define CONV_DIM_ (DS_ + 2*G_*S_)      // 4352
#define PROJ_     (2*DS_ + 2*G_*S_ + NH_) // 8512
#define L_        1024
#define EPS_      1e-5f
#define DT_OFF_   (2*DS_ + 2*G_*S_)    // 8448

typedef __bf16 bf16;
typedef __attribute__((ext_vector_type(16))) __bf16 v16bf;
typedef __attribute__((ext_vector_type(8)))  __bf16 v8bf;
typedef __attribute__((ext_vector_type(8)))  float  v8f;

// ---------------- small helpers ----------------
__device__ __forceinline__ float silu_f(float x) { return x / (1.0f + __expf(-x)); }
__device__ __forceinline__ float softplus_f(float x) {
  return (x > 20.0f) ? x : log1pf(__expf(x));
}

// Async 16-byte global -> LDS copy (gfx1250 ASYNCcnt path).
// lds_off = LDS byte offset (low 32 bits of generic shared pointer per ISA
// flat-aperture rule); gaddr = 64-bit global address.
__device__ __forceinline__ void async_g2l_b128(uint32_t lds_off, const void* gaddr) {
  asm volatile("global_load_async_to_lds_b128 %0, %1, off"
               :: "v"(lds_off), "v"((uint64_t)(uintptr_t)gaddr)
               : "memory");
}
__device__ __forceinline__ void wait_asynccnt0() {
  asm volatile("s_wait_asynccnt 0" ::: "memory");
}

__device__ __forceinline__ float block_reduce_sum(float v, float* red) {
  #pragma unroll
  for (int o = 16; o > 0; o >>= 1) v += __shfl_xor(v, o, 32);
  const int wave = threadIdx.x >> 5, lane = threadIdx.x & 31;
  if (lane == 0) red[wave] = v;
  __syncthreads();
  float s = (threadIdx.x < 8) ? red[threadIdx.x] : 0.0f;
  if (wave == 0) {
    #pragma unroll
    for (int o = 4; o > 0; o >>= 1) s += __shfl_xor(s, o, 32);
    if (lane == 0) red[0] = s;
  }
  __syncthreads();
  return red[0];
}

// ---------------- elementwise converts / copies ----------------
__global__ void f32_to_bf16_kernel(const float* __restrict__ in, bf16* __restrict__ out, size_t n) {
  size_t i = (size_t)blockIdx.x * blockDim.x + threadIdx.x;
  const size_t stride = (size_t)gridDim.x * blockDim.x;
  for (; i < n; i += stride) out[i] = (bf16)in[i];
}

__global__ void copy_f32_kernel(const float* __restrict__ in, float* __restrict__ out, size_t n) {
  size_t i = (size_t)blockIdx.x * blockDim.x + threadIdx.x;
  const size_t stride = (size_t)gridDim.x * blockDim.x;
  for (; i < n; i += stride) out[i] = in[i];
}

// ---------------- input RMSNorm -> bf16 ----------------
__global__ __launch_bounds__(256)
void rmsnorm_bf16_kernel(const float* __restrict__ x, const float* __restrict__ w,
                         bf16* __restrict__ out) {
  __shared__ float red[8];
  const int row = blockIdx.x;
  const float* xr = x + (size_t)row * H_;
  float ss = 0.0f;
  for (int i = threadIdx.x; i < H_; i += 256) { float v = xr[i]; ss += v * v; }
  const float tot = block_reduce_sum(ss, red);
  const float inv = rsqrtf(tot / (float)H_ + EPS_);
  bf16* orow = out + (size_t)row * H_;
  for (int i = threadIdx.x; i < H_; i += 256) orow[i] = (bf16)(xr[i] * inv * w[i]);
}

// ---------------- gated RMSNorm (y * silu(z)) -> bf16 ----------------
__global__ __launch_bounds__(256)
void gated_rmsnorm_bf16_kernel(const float* __restrict__ y, const float* __restrict__ zx,
                               const float* __restrict__ w, bf16* __restrict__ out) {
  __shared__ float red[8];
  const int row = blockIdx.x;
  const float* yr = y + (size_t)row * DS_;
  const float* zr = zx + (size_t)row * PROJ_;   // z = first DS_ columns of zxbcdt
  float ss = 0.0f;
  for (int i = threadIdx.x; i < DS_; i += 256) {
    float g = yr[i] * silu_f(zr[i]);
    ss += g * g;
  }
  const float inv = rsqrtf(block_reduce_sum(ss, red) / (float)DS_ + EPS_);
  bf16* orow = out + (size_t)row * DS_;
  for (int i = threadIdx.x; i < DS_; i += 256) {
    float g = yr[i] * silu_f(zr[i]);
    orow[i] = (bf16)(g * inv * w[i]);
  }
}

// ---------------- causal depthwise conv (K=4) + SiLU ----------------
__global__ __launch_bounds__(256)
void conv_silu_kernel(const float* __restrict__ zx, const float* __restrict__ conv_w,
                      float* __restrict__ xbc) {
  const int idx = blockIdx.x * blockDim.x + threadIdx.x;   // over L_*CONV_DIM_
  if (idx >= L_ * CONV_DIM_) return;
  const int c = idx % CONV_DIM_;
  const int t = idx / CONV_DIM_;
  float s = 0.0f;
  #pragma unroll
  for (int k = 0; k < KCONV_; ++k) {
    const int tt = t + k - (KCONV_ - 1);
    if (tt >= 0) s += zx[(size_t)tt * PROJ_ + DS_ + c] * conv_w[c * KCONV_ + k];
  }
  xbc[idx] = silu_f(s);
}

// ---------------- selective scan: one workgroup per head ----------------
__global__ __launch_bounds__(256)
void ssm_scan_kernel(const float* __restrict__ xbc, const float* __restrict__ zx,
                     const float* __restrict__ A_log, const float* __restrict__ dt_bias,
                     const float* __restrict__ Dv, float* __restrict__ y) {
  __shared__ float sX[HD_];
  __shared__ float sB[S_];
  __shared__ float sC[S_];
  const int hd   = blockIdx.x;       // head 0..63
  const int tid  = threadIdx.x;
  const int p    = tid >> 2;         // head_dim index 0..63
  const int nseg = tid & 3;          // 32-wide state segment
  const int n0   = nseg * 32;
  const float A   = -__expf(A_log[hd]);
  const float Dh  = Dv[hd];
  const float dtb = dt_bias[hd];
  float h[32];
  #pragma unroll
  for (int j = 0; j < 32; ++j) h[j] = 0.0f;

  for (int t = 0; t < L_; ++t) {
    const float* xr = xbc + (size_t)t * CONV_DIM_;
    for (int i = tid; i < HD_ + 2 * S_; i += 256) {
      if      (i < HD_)        sX[i]             = xr[hd * HD_ + i];
      else if (i < HD_ + S_)   sB[i - HD_]       = xr[DS_ + (i - HD_)];
      else                     sC[i - HD_ - S_]  = xr[DS_ + S_ + (i - HD_ - S_)];
    }
    const float dtraw = zx[(size_t)t * PROJ_ + DT_OFF_ + hd] + dtb;  // uniform -> scalar load
    const float dt = softplus_f(dtraw);
    const float dA = __expf(dt * A);
    __syncthreads();
    const float xp = sX[p];
    const float coef = dt * xp;
    float part = 0.0f;
    #pragma unroll
    for (int j = 0; j < 32; ++j) {
      h[j] = h[j] * dA + coef * sB[n0 + j];
      part += h[j] * sC[n0 + j];
    }
    part += __shfl_xor(part, 1, 32);
    part += __shfl_xor(part, 2, 32);
    if (nseg == 0)
      y[(size_t)t * DS_ + hd * HD_ + p] = part + Dh * xp;
    __syncthreads();
  }
}

// ---------------- bf16 WMMA GEMM: C[M,N] = A[M,K] x B[K,N] (* colScale) ----------------
// Block tile 128(M) x 64(N), K-slab 32. 8 wave32 per block, each wave owns 32x32
// (2x2 grid of v_wmma_f32_16x16x32_bf16 accumulators). Requires M%128==0, N%64==0, K%32==0.
// A tile staged with global_load_async_to_lds_b128 (ASYNCcnt); B tile staged
// pre-swizzled into the WMMA B-fragment layout so fragment loads are ds_load_b128.
__global__ __launch_bounds__(256)
void wmma_gemm_bf16(const bf16* __restrict__ A, const bf16* __restrict__ B,
                    float* __restrict__ C, const float* __restrict__ colScale,
                    int M, int N, int K) {
  __shared__ __align__(16) bf16 As[128][40];        // 128x32 tile, padded stride (80B rows)
  __shared__ __align__(32) bf16 Bsw[4][32][16];     // [colGroup][lane][elem] fragment-major
  const int tid  = threadIdx.x;
  const int lane = tid & 31;
  const int wave = tid >> 5;
  const int m0 = blockIdx.y * 128;
  const int n0 = blockIdx.x * 64;
  const int mw = (wave >> 1) * 32;   // wave row offset within block tile
  const int nw = (wave & 1) * 32;    // wave col offset within block tile

  const int arow   = tid >> 1;          // 0..127
  const int achunk = (tid & 1) * 16;    // 0 or 16
  const int brow   = tid >> 3;          // k within slab: 0..31
  const int bchunk = (tid & 7) * 8;     // col base: 0..56

  // Inverse of ISA 7.12.2 16-bit B(32x16) layout for the staging scatter:
  // lane = (col&15) + 16*hi,  hi = (k>>3)&1,  elem = k'' or k''-8,
  // k'' = (k&7) + 16*(k>=16).
  const int bhi  = (brow >> 3) & 1;
  const int kpp  = (brow & 7) + (brow & 16);
  const int belm = (kpp < 8) ? kpp : (kpp - 8);

  v8f acc[2][2] = {};

  for (int k0 = 0; k0 < K; k0 += 32) {
    __syncthreads();
    {   // A tile: 128x32 bf16 via async global->LDS (two b128 per thread)
      const bf16* gp = A + (size_t)(m0 + arow) * K + k0 + achunk;
      async_g2l_b128((uint32_t)(uintptr_t)&As[arow][achunk],     gp);
      async_g2l_b128((uint32_t)(uintptr_t)&As[arow][achunk + 8], gp + 8);
    }
    {   // B tile: 32x64 bf16, vector load + fragment-major scatter into LDS
      const v8bf bv = *(const v8bf*)(B + (size_t)(k0 + brow) * N + n0 + bchunk);
      #pragma unroll
      for (int j = 0; j < 8; ++j) {
        const int c = bchunk + j;
        Bsw[c >> 4][(c & 15) + 16 * bhi][belm] = bv[j];
      }
    }
    if (k0 + 32 < K) {  // gfx1250 global_prefetch_b8 on next K-slab
      __builtin_prefetch(A + (size_t)(m0 + arow) * K + (k0 + 32) + achunk, 0, 1);
      __builtin_prefetch(B + (size_t)(k0 + 32 + brow) * N + n0 + bchunk, 0, 1);
    }
    wait_asynccnt0();
    __syncthreads();

    // A fragments per ISA 7.12.2 (contiguous 16B chunks -> ds_load_b128 x2).
    v16bf af[2], bfr[2];
    const int l16 = lane & 15;
    const int hi8 = (lane >> 4) * 8;
    #pragma unroll
    for (int mi = 0; mi < 2; ++mi) {
      const int rowb = mw + mi * 16 + l16;
      #pragma unroll
      for (int r = 0; r < 8; ++r) {
        const int kk = ((r < 4) ? (2 * r) : (16 + 2 * (r - 4))) + hi8;
        af[mi][2 * r]     = As[rowb][kk];
        af[mi][2 * r + 1] = As[rowb][kk + 1];
      }
    }
    // B fragments: one contiguous 32B LDS read per lane.
    #pragma unroll
    for (int ni = 0; ni < 2; ++ni)
      bfr[ni] = *(const v16bf*)&Bsw[(nw >> 4) + ni][lane][0];

    #pragma unroll
    for (int mi = 0; mi < 2; ++mi)
      #pragma unroll
      for (int ni = 0; ni < 2; ++ni)
        acc[mi][ni] = __builtin_amdgcn_wmma_f32_16x16x32_bf16(
            false, af[mi], false, bfr[ni], (short)0, acc[mi][ni], false, false);
  }

  // Store D: lane l holds column l%16; VGPR r holds row r + 8*(l>=16).
  const int l16 = lane & 15;
  const int hi8 = (lane >> 4) * 8;
  #pragma unroll
  for (int mi = 0; mi < 2; ++mi) {
    #pragma unroll
    for (int ni = 0; ni < 2; ++ni) {
      const int col = n0 + nw + ni * 16 + l16;
      const float sc = colScale ? colScale[col] : 1.0f;
      const int rb = m0 + mw + mi * 16 + hi8;
      #pragma unroll
      for (int r = 0; r < 8; ++r)
        C[(size_t)(rb + r) * N + col] = acc[mi][ni][r] * sc;
    }
  }
}

// ---------------- launcher ----------------
extern "C" void kernel_launch(void* const* d_in, const int* in_sizes, int n_in,
                              void* d_out, int out_size, void* d_ws, size_t ws_size,
                              hipStream_t stream) {
  (void)in_sizes; (void)n_in; (void)out_size; (void)ws_size;
  const float* hidden     = (const float*)d_in[0];
  const float* residual   = (const float*)d_in[1];
  const float* ln_w       = (const float*)d_in[2];
  const float* in_proj_w  = (const float*)d_in[3];
  const float* conv_w     = (const float*)d_in[4];
  const float* A_log      = (const float*)d_in[5];
  const float* dt_bias    = (const float*)d_in[6];
  const float* Dv         = (const float*)d_in[7];
  const float* norm_w     = (const float*)d_in[8];
  const float* out_proj_w = (const float*)d_in[9];
  const float* mup        = (const float*)d_in[10];

  uint8_t* ws = (uint8_t*)d_ws;
  size_t off = 0;
  auto wsalloc = [&](size_t bytes) -> void* {
    void* p = ws + off;
    off = (off + bytes + 255) & ~(size_t)255;
    return p;
  };
  bf16*  w1b = (bf16*) wsalloc((size_t)H_  * PROJ_     * sizeof(bf16));
  bf16*  w2b = (bf16*) wsalloc((size_t)DS_ * H_        * sizeof(bf16));
  bf16*  xnb = (bf16*) wsalloc((size_t)L_  * H_        * sizeof(bf16));
  float* zx  = (float*)wsalloc((size_t)L_  * PROJ_     * sizeof(float));
  float* xbc = (float*)wsalloc((size_t)L_  * CONV_DIM_ * sizeof(float));
  float* yf  = (float*)wsalloc((size_t)L_  * DS_       * sizeof(float));
  bf16*  ybb = (bf16*) wsalloc((size_t)L_  * DS_       * sizeof(bf16));

  // 1) fp32 -> bf16 weight casts (fresh every call; no cross-call state)
  f32_to_bf16_kernel<<<2048, 256, 0, stream>>>(in_proj_w,  w1b, (size_t)H_ * PROJ_);
  f32_to_bf16_kernel<<<2048, 256, 0, stream>>>(out_proj_w, w2b, (size_t)DS_ * H_);

  // 2) input RMSNorm -> bf16 activations
  rmsnorm_bf16_kernel<<<L_, 256, 0, stream>>>(hidden, ln_w, xnb);

  // 3) GEMM1: zxbcdt = (xn @ in_proj_w) * mup   [1024 x 8512]
  wmma_gemm_bf16<<<dim3(PROJ_ / 64, L_ / 128), 256, 0, stream>>>(
      xnb, w1b, zx, mup, L_, PROJ_, H_);

  // 4) causal depthwise conv + SiLU on xBC slice
  {
    const int tot = L_ * CONV_DIM_;
    conv_silu_kernel<<<(tot + 255) / 256, 256, 0, stream>>>(zx, conv_w, xbc);
  }

  // 5) selective scan (one workgroup per head), y includes +D*x
  ssm_scan_kernel<<<NH_, 256, 0, stream>>>(xbc, zx, A_log, dt_bias, Dv, yf);

  // 6) gated RMSNorm -> bf16
  gated_rmsnorm_bf16_kernel<<<L_, 256, 0, stream>>>(yf, zx, norm_w, ybb);

  // 7) GEMM2: out = yb @ out_proj_w  [1024 x 4096] -> first half of d_out (fp32)
  wmma_gemm_bf16<<<dim3(H_ / 64, L_ / 128), 256, 0, stream>>>(
      ybb, w2b, (float*)d_out, nullptr, L_, H_, DS_);

  // 8) residual passthrough -> second half of d_out
  copy_f32_kernel<<<2048, 256, 0, stream>>>(
      residual, (float*)d_out + (size_t)L_ * H_, (size_t)L_ * H_);
}
